// SRNN_57080115363998
// MI455X (gfx1250) — matrix operand order repaired and checked by reading, not compiled, not run on hardware
//
#include <hip/hip_runtime.h>

// ---------------- problem constants ----------------
constexpr int T = 100, B = 256, D = 700, H = 1024, O = 20;
constexpr int DP = 704;                 // D padded to multiple of 32 (K tile)
constexpr int TB = T * B;               // 25600 rows of the hoisted input GEMM
constexpr float THR   = 1.0f;
constexpr float ALPHA = 0.9512294245007140f;   // exp(-1/20)
constexpr float KAPPA = 0.9512294245007140f;   // exp(-1/20)
constexpr int GRID_STEPS = 32;          // persistent-kernel blocks (8 waves each)

// ---------------- workspace layout (bytes, all 256B aligned) ----------------
constexpr size_t OFF_XH    = 0;                                   // f16 [TB][DP]
constexpr size_t SZ_XH     = (size_t)TB * DP * 2;
constexpr size_t OFF_WINT  = OFF_XH + SZ_XH;                      // f16 [DP][H]
constexpr size_t SZ_WINT   = (size_t)DP * H * 2;
constexpr size_t OFF_WRECT = OFF_WINT + SZ_WINT;                  // f16 [H][H] transposed, diag=0
constexpr size_t SZ_WRECT  = (size_t)H * H * 2;
constexpr size_t OFF_AIN   = OFF_WRECT + SZ_WRECT;                // f32 [TB][H]
constexpr size_t SZ_AIN    = (size_t)TB * H * 4;
constexpr size_t OFF_V     = OFF_AIN + SZ_AIN;                    // f32 [B][H]
constexpr size_t SZ_V      = (size_t)B * H * 4;
constexpr size_t OFF_Z     = OFF_V + SZ_V;                        // f16 [2][B][H] ping-pong
constexpr size_t SZ_Z      = (size_t)2 * B * H * 2;
constexpr size_t OFF_VO    = OFF_Z + SZ_Z;                        // f32 [B][O]
constexpr size_t SZ_VO     = 20480;
constexpr size_t OFF_SYNC  = OFF_VO + SZ_VO;                      // u32 grid-barrier counter

// ---------------- vector types ----------------
typedef _Float16 half8 __attribute__((ext_vector_type(8)));
typedef _Float16 v16h  __attribute__((ext_vector_type(16)));
typedef float    v8f   __attribute__((ext_vector_type(8)));

// ---------------- fragment loaders (wave32, 16-bit WMMA layouts) ----------------
// A 16x32 f16: lane<16 holds row m0+(l&15), K = {k0..k0+7, k0+16..k0+23};
//              lane>=16 same row,           K = {k0+8..k0+15, k0+24..k0+31}.
__device__ __forceinline__ v16h load_a_frag(const _Float16* __restrict__ base,
                                            int rowStride, int m0, int k0, int lane) {
  int m  = m0 + (lane & 15);
  int kb = k0 + ((lane & 16) ? 8 : 0);
  const _Float16* p = base + (long)m * rowStride + kb;
  half8 lo = *(const half8*)(p);
  half8 hi = *(const half8*)(p + 16);
  return __builtin_shufflevector(lo, hi, 0,1,2,3,4,5,6,7,8,9,10,11,12,13,14,15);
}

// B 32x16 f16: lane l holds column N=n0+(l&15), K = {k0+8h..k0+8h+7, k0+16+8h..k0+16+8h+7}, h=l>>4.
__device__ __forceinline__ v16h load_b_frag(const _Float16* __restrict__ base,
                                            int rowStride, int k0, int n0, int lane) {
  int kb = k0 + ((lane & 16) ? 8 : 0);
  const _Float16* p = base + (long)kb * rowStride + n0 + (lane & 15);
  v16h r;
#pragma unroll
  for (int s = 0; s < 8; ++s) {
    r[s]     = p[(long)s * rowStride];
    r[s + 8] = p[(long)(s + 16) * rowStride];
  }
  return r;
}

#define WMMA_F16(A, Bm, C) \
  __builtin_amdgcn_wmma_f32_16x16x32_f16(false, (A), false, (Bm), (short)0, (C), false, false)

// ---------------- device-wide barrier (monotonic counter, no reset) ----------------
__device__ __forceinline__ void grid_sync(unsigned* c, unsigned target) {
  __builtin_amdgcn_fence(__ATOMIC_RELEASE, "agent");
  __syncthreads();
  if (threadIdx.x == 0) {
    __hip_atomic_fetch_add(c, 1u, __ATOMIC_RELAXED, __HIP_MEMORY_SCOPE_AGENT);
    while (__hip_atomic_load(c, __ATOMIC_RELAXED, __HIP_MEMORY_SCOPE_AGENT) < target)
      __builtin_amdgcn_s_sleep(1);
  }
  __syncthreads();
  __builtin_amdgcn_fence(__ATOMIC_ACQUIRE, "agent");
}

// ---------------- prep: transpose/convert weights to f16 ----------------
__global__ void __launch_bounds__(256) k_prep_w(const float* __restrict__ w_in,
                                                const float* __restrict__ w_rec,
                                                _Float16* __restrict__ w_inT,
                                                _Float16* __restrict__ w_recT) {
  int idx = blockIdx.x * 256 + threadIdx.x;
  if (idx < DP * H) {                       // w_inT[d][h] = w_in[h][d], pad d>=700 with 0
    int k = idx / H, n = idx % H;
    float val = (k < D) ? w_in[(long)n * D + k] : 0.0f;
    w_inT[idx] = (_Float16)val;
  }
  int idx2 = idx - DP * H;
  if (idx2 >= 0 && idx2 < H * H) {          // w_recT[j][i] = (i==j) ? 0 : w_rec[i][j]
    int j = idx2 / H, i = idx2 % H;
    float val = (i == j) ? 0.0f : w_rec[(long)i * H + j];
    w_recT[idx2] = (_Float16)val;
  }
}

// ---------------- prep: x -> f16, K-padded ----------------
__global__ void __launch_bounds__(256) k_prep_x(const float* __restrict__ x,
                                                _Float16* __restrict__ xh) {
  long idx = (long)blockIdx.x * 256 + threadIdx.x;
  if (idx >= (long)TB * DP) return;
  int row = (int)(idx / DP), d = (int)(idx % DP);
  xh[idx] = (d < D) ? (_Float16)x[(long)row * D + d] : (_Float16)0.0f;
}

// ---------------- init state, out[0] = softmax(0), barrier counter ----------------
__global__ void __launch_bounds__(256) k_init(float* __restrict__ v,
                                              _Float16* __restrict__ zbuf,
                                              float* __restrict__ vo,
                                              float* __restrict__ out0,
                                              unsigned* __restrict__ sync) {
  int idx = blockIdx.x * 256 + threadIdx.x;
  if (idx < B * H) {
    v[idx] = 0.0f;
    zbuf[idx] = (_Float16)0.0f;
    zbuf[B * H + idx] = (_Float16)0.0f;
  }
  if (idx < B * O) {
    vo[idx] = 0.0f;
    out0[idx] = 1.0f / (float)O;
  }
  if (idx == 0) *sync = 0u;
}

// ---------------- hoisted input GEMM: A_in[TB][H] = xh @ w_inT (double-buffered) ----------------
__global__ void __launch_bounds__(256) k_gemm_in(const _Float16* __restrict__ xh,
                                                 const _Float16* __restrict__ w_inT,
                                                 float* __restrict__ A_in) {
  int wave = blockIdx.x * 8 + (threadIdx.x >> 5);   // 25600 waves total
  int lane = threadIdx.x & 31;
  int m0 = (wave >> 4) * 16;                        // 1600 M tiles
  int n0 = (wave & 15) * 64;                        // 16 N chunks of 64
  v8f c0 = {}, c1 = {}, c2 = {}, c3 = {};
  constexpr int NK = DP / 32;

  v16h a  = load_a_frag(xh, DP, m0, 0, lane);
  v16h b0 = load_b_frag(w_inT, H, 0, n0,      lane);
  v16h b1 = load_b_frag(w_inT, H, 0, n0 + 16, lane);
  v16h b2 = load_b_frag(w_inT, H, 0, n0 + 32, lane);
  v16h b3 = load_b_frag(w_inT, H, 0, n0 + 48, lane);
  for (int kt = 0; kt < NK; ++kt) {
    int kn = (kt + 1 < NK) ? (kt + 1) * 32 : kt * 32;    // last iter: benign reload
    __builtin_prefetch(xh + (long)(m0 + (lane & 15)) * DP + kn, 0, 3);
    v16h an  = load_a_frag(xh, DP, m0, kn, lane);
    v16h bn0 = load_b_frag(w_inT, H, kn, n0,      lane);
    v16h bn1 = load_b_frag(w_inT, H, kn, n0 + 16, lane);
    v16h bn2 = load_b_frag(w_inT, H, kn, n0 + 32, lane);
    v16h bn3 = load_b_frag(w_inT, H, kn, n0 + 48, lane);
    c0 = WMMA_F16(a, b0, c0);
    c1 = WMMA_F16(a, b1, c1);
    c2 = WMMA_F16(a, b2, c2);
    c3 = WMMA_F16(a, b3, c3);
    a = an; b0 = bn0; b1 = bn1; b2 = bn2; b3 = bn3;
  }
  int mrow = m0 + 8 * ((lane >> 4) & 1);
  int ncol = n0 + (lane & 15);
#pragma unroll
  for (int r = 0; r < 8; ++r) {
    float* p = A_in + (long)(mrow + r) * H + ncol;
    p[0]  = c0[r];
    p[16] = c1[r];
    p[32] = c2[r];
    p[48] = c3[r];
  }
}

// ---------------- persistent kernel: 99 LIF steps, one launch ----------------
__global__ void __launch_bounds__(256) k_steps(_Float16* __restrict__ zbuf,
                                               float* __restrict__ v,
                                               const float* __restrict__ A_in,
                                               const _Float16* __restrict__ w_recT,
                                               const float* __restrict__ w_out,
                                               float* __restrict__ vo,
                                               float* __restrict__ out,
                                               unsigned* __restrict__ sync) {
  const int wIdx = threadIdx.x >> 5;
  const int lane = threadIdx.x & 31;
  const int wave = blockIdx.x * 8 + wIdx;           // 256 waves total
  const int m0 = (wave >> 4) * 16;                  // 16 M tiles (B=256)
  const int n0 = (wave & 15) * 64;                  // 16 N chunks (H=1024)
  const int row = blockIdx.x * 8 + wIdx;            // readout batch row (256 rows)
  const int oc = (lane < O) ? lane : (O - 1);
  __shared__ float s[8][O];

  for (int t = 0; t < T - 1; ++t) {
    const _Float16* zold = zbuf + (size_t)(t & 1) * (B * H);
    _Float16*       znew = zbuf + (size_t)((t + 1) & 1) * (B * H);
    const float*    A_t  = A_in + (size_t)t * (B * H);

    // ---- recurrent GEMM (double-buffered fragments) ----
    v8f c0 = {}, c1 = {}, c2 = {}, c3 = {};
    v16h a  = load_a_frag(zold, H, m0, 0, lane);
    v16h b0 = load_b_frag(w_recT, H, 0, n0,      lane);
    v16h b1 = load_b_frag(w_recT, H, 0, n0 + 16, lane);
    v16h b2 = load_b_frag(w_recT, H, 0, n0 + 32, lane);
    v16h b3 = load_b_frag(w_recT, H, 0, n0 + 48, lane);
    constexpr int NK = H / 32;
    for (int kt = 0; kt < NK; ++kt) {
      int kn = (kt + 1 < NK) ? (kt + 1) * 32 : kt * 32;  // last iter: benign reload
      v16h an  = load_a_frag(zold, H, m0, kn, lane);
      v16h bn0 = load_b_frag(w_recT, H, kn, n0,      lane);
      v16h bn1 = load_b_frag(w_recT, H, kn, n0 + 16, lane);
      v16h bn2 = load_b_frag(w_recT, H, kn, n0 + 32, lane);
      v16h bn3 = load_b_frag(w_recT, H, kn, n0 + 48, lane);
      c0 = WMMA_F16(a, b0, c0);
      c1 = WMMA_F16(a, b1, c1);
      c2 = WMMA_F16(a, b2, c2);
      c3 = WMMA_F16(a, b3, c3);
      a = an; b0 = bn0; b1 = bn1; b2 = bn2; b3 = bn3;
    }

    // ---- LIF epilogue: v update + spike ----
    int mrow = m0 + 8 * ((lane >> 4) & 1);
    int ncol = n0 + (lane & 15);
#pragma unroll
    for (int r = 0; r < 8; ++r) {
      int m = mrow + r;
#pragma unroll
      for (int nt = 0; nt < 4; ++nt) {
        float rec = (nt == 0) ? c0[r] : (nt == 1) ? c1[r] : (nt == 2) ? c2[r] : c3[r];
        long idx = (long)m * H + ncol + nt * 16;
        float zo = (float)zold[idx];
        float vn = ALPHA * v[idx] + rec + A_t[idx] - THR * zo;
        v[idx] = vn;
        znew[idx] = (_Float16)((vn > THR) ? 1.0f : 0.0f);
      }
    }

    // ---- device-wide barrier: znew complete before anyone consumes it ----
    grid_sync(sync, (unsigned)((t + 1) * GRID_STEPS));

    // ---- readout: vo = kappa*vo + znew @ w_outT, softmax -> out[t+1] ----
    {
      const _Float16* zr = znew + (size_t)row * H;
      const float*    wr = w_out + (size_t)oc * H;
      float acc = 0.0f;
#pragma unroll 4
      for (int j = 0; j < H; j += 8) {
        half8 zv = *(const half8*)(zr + j);
        const float4* wp = (const float4*)(wr + j);
        float4 w0 = wp[0], w1 = wp[1];
        acc += (float)zv[0] * w0.x + (float)zv[1] * w0.y +
               (float)zv[2] * w0.z + (float)zv[3] * w0.w +
               (float)zv[4] * w1.x + (float)zv[5] * w1.y +
               (float)zv[6] * w1.z + (float)zv[7] * w1.w;
      }
      float vn = KAPPA * vo[row * O + oc] + acc;
      if (lane < O) { vo[row * O + lane] = vn; s[wIdx][lane] = vn; }
    }
    __syncthreads();
    if (lane < O) {
      float mx = s[wIdx][0];
#pragma unroll
      for (int i = 1; i < O; ++i) mx = fmaxf(mx, s[wIdx][i]);
      float sum = 0.0f;
#pragma unroll
      for (int i = 0; i < O; ++i) sum += __expf(s[wIdx][i] - mx);
      out[(size_t)(t + 1) * B * O + row * O + lane] = __expf(s[wIdx][lane] - mx) / sum;
    }
    __syncthreads();   // protect s[] before next step rewrites it
  }
}

// ---------------- launcher ----------------
extern "C" void kernel_launch(void* const* d_in, const int* in_sizes, int n_in,
                              void* d_out, int out_size, void* d_ws, size_t ws_size,
                              hipStream_t stream) {
  const float* x     = (const float*)d_in[0];   // [T,B,D]
  const float* w_in  = (const float*)d_in[1];   // [H,D]
  const float* w_rec = (const float*)d_in[2];   // [H,H]
  const float* w_out = (const float*)d_in[3];   // [O,H]
  float* out = (float*)d_out;                   // [T,B,O]

  char* ws = (char*)d_ws;
  _Float16* xh     = (_Float16*)(ws + OFF_XH);
  _Float16* w_inT  = (_Float16*)(ws + OFF_WINT);
  _Float16* w_recT = (_Float16*)(ws + OFF_WRECT);
  float*    A_in   = (float*)   (ws + OFF_AIN);
  float*    v      = (float*)   (ws + OFF_V);
  _Float16* zbuf   = (_Float16*)(ws + OFF_Z);
  float*    vo     = (float*)   (ws + OFF_VO);
  unsigned* sync   = (unsigned*)(ws + OFF_SYNC);

  // 1) weight transpose/convert
  k_prep_w<<<(DP * H + H * H) / 256, 256, 0, stream>>>(w_in, w_rec, w_inT, w_recT);
  // 2) x -> f16 padded
  k_prep_x<<<(int)(((long)TB * DP) / 256), 256, 0, stream>>>(x, xh);
  // 3) zero state, out[0] = softmax(0), reset barrier counter
  k_init<<<(B * H) / 256, 256, 0, stream>>>(v, zbuf, vo, out, sync);
  // 4) hoisted input GEMM over all T (25600 waves, 8 per block)
  k_gemm_in<<<TB / 8, 256, 0, stream>>>(xh, w_inT, A_in);
  // 5) all 99 sequential LIF steps in ONE persistent kernel (32 co-resident blocks)
  k_steps<<<GRID_STEPS, 256, 0, stream>>>(zbuf, v, A_in, w_recT, w_out, vo, out, sync);
}
